// Graph_noSub_75196287418589
// MI455X (gfx1250) — matrix operand order repaired and checked by reading, not compiled, not run on hardware
//
#include <hip/hip_runtime.h>
#include <math.h>

// ---------------- WMMA types (CDNA5 / gfx1250, wave32) ----------------
typedef __attribute__((ext_vector_type(16))) _Float16 v16h;
typedef __attribute__((ext_vector_type(8)))  _Float16 v8h_t;   // 16 bytes
typedef __attribute__((ext_vector_type(8)))  float    v8f;

// ---------------- problem constants (from reference) ----------------
#define N_GRAPHS 64
#define MOL_NPG  40
#define PRO_NPG  500
#define NM   (N_GRAPHS * MOL_NPG)   // 2560
#define NPN  (N_GRAPHS * PRO_NPG)   // 32000
#define EM   (NM * 4)               // 10240
#define EPN  (NPN * 8)              // 256000
#define FM   78
#define DMH  312
#define FP   33
#define DPH  132
#define HH   2
#define OUTF 128

// ---------------- helpers ----------------
__device__ __forceinline__ unsigned f2ord(float f) {
    unsigned u = __float_as_uint(f);
    return (u & 0x80000000u) ? ~u : (u | 0x80000000u);
}
__device__ __forceinline__ float ord2f(unsigned u) {
    return __uint_as_float((u & 0x80000000u) ? (u & 0x7FFFFFFFu) : ~u);
}
__device__ __forceinline__ float lrelu02(float v) { return v > 0.f ? v : 0.2f * v; }

// =====================================================================
// LDS-staged WMMA GEMM:  C[M,N] = act(A[M,K] @ B[K,N] + bias)
//
// Block = 256 threads = 8 wave32; macro tile 64(M) x 32(N); each wave
// owns one 16x16 tile; K-loop step 32 with fp32->fp16 staging into LDS.
// Interior tiles take a branch-free staging fast path (block-uniform
// bounds test); edge tiles use clamped/guarded staging. Per-lane
// fragments are two ds_load_b128 each, matching the CDNA5 ISA 7.12.2
// A/B register layouts. act: 0=none 1=relu 2=tanh
// =====================================================================
#define TM 64
#define TN 32
#define TK 32
#define LDS_STR 40   // f16 stride; 40*2 = 80 bytes, 16B-aligned rows

__global__ __launch_bounds__(256)
void wmma_gemm_kernel(const float* __restrict__ A,
                      const float* __restrict__ B,
                      const float* __restrict__ bias,
                      float* __restrict__ C,
                      int M, int N, int K, int act)
{
    __shared__ __align__(16) _Float16 Ash[TM * LDS_STR];
    __shared__ __align__(16) _Float16 Bsh[TN * LDS_STR];

    const int tid  = threadIdx.x;
    const int lane = tid & 31;
    const int wave = tid >> 5;            // 0..7
    const int half = lane >> 4;           // lane group
    const int r    = lane & 15;

    const int m0 = blockIdx.y * TM;
    const int n0 = blockIdx.x * TN;
    const int tm = (wave >> 1) * 16;      // wave tile offset in macro tile
    const int tn = (wave & 1) * 16;

    const bool fullM = (m0 + TM <= M);
    const bool fullN = (n0 + TN <= N);

    // per-thread staging coordinates (fixed across K-steps)
    const int a_m = tid >> 5, a_k = tid & 31;       // A: 8 rows apart per step
    const int b_k = tid >> 5, b_n = tid & 31;       // B: 8 k-rows apart per step

    v8f acc = {};

    for (int k0 = 0; k0 < K; k0 += TK) {
        const bool fullK = (k0 + TK <= K);

        if (fullM && fullK) {              // ---- branch-free A staging ----
#pragma unroll
            for (int i = 0; i < 8; ++i) {
                int mm = a_m + i * 8;
                Ash[mm * LDS_STR + a_k] =
                    (_Float16)A[(size_t)(m0 + mm) * K + (k0 + a_k)];
            }
        } else {                           // ---- guarded A staging ----
#pragma unroll
            for (int i = 0; i < 8; ++i) {
                int mm = a_m + i * 8;
                int gm = m0 + mm, gk = k0 + a_k;
                float v = A[(size_t)min(gm, M - 1) * K + min(gk, K - 1)];
                if (gm >= M || gk >= K) v = 0.f;
                Ash[mm * LDS_STR + a_k] = (_Float16)v;
            }
        }
        if (fullN && fullK) {              // ---- branch-free B staging ----
#pragma unroll
            for (int i = 0; i < 4; ++i) {
                int kk = b_k + i * 8;
                Bsh[b_n * LDS_STR + kk] =
                    (_Float16)B[(size_t)(k0 + kk) * N + (n0 + b_n)];
            }
        } else {                           // ---- guarded B staging ----
#pragma unroll
            for (int i = 0; i < 4; ++i) {
                int kk = b_k + i * 8;
                int gk = k0 + kk, gn = n0 + b_n;
                float v = B[(size_t)min(gk, K - 1) * N + min(gn, N - 1)];
                if (gk >= K || gn >= N) v = 0.f;
                Bsh[b_n * LDS_STR + kk] = (_Float16)v;
            }
        }
        __syncthreads();

        // ---- per-lane fragments: two 16B LDS reads each ----
        v16h a, b;
        {
            const _Float16* ar = &Ash[(tm + r) * LDS_STR];
            v8h_t alo = *(const v8h_t*)(ar + 8 * half);        // K = 8h .. 8h+7
            v8h_t ahi = *(const v8h_t*)(ar + 16 + 8 * half);   // K = 16+8h ..
            const _Float16* br = &Bsh[(tn + r) * LDS_STR];
            v8h_t blo = *(const v8h_t*)(br + 8 * half);        // K = 8h .. 8h+7
            v8h_t bhi = *(const v8h_t*)(br + 16 + 8 * half);   // K = 16+8h ..
#pragma unroll
            for (int j = 0; j < 8; ++j) {
                a[j] = alo[j]; a[8 + j] = ahi[j];
                b[j] = blo[j]; b[8 + j] = bhi[j];
            }
        }
        acc = __builtin_amdgcn_wmma_f32_16x16x32_f16(false, a, false, b,
                                                     (short)0, acc, false, false);
        __syncthreads();
    }

    // C/D layout: VGPR v holds M = tile + v + 8*half, N = tile + (lane%16)
    const int n = n0 + tn + r;
    if (n < N) {
        float bv = bias ? bias[n] : 0.f;
#pragma unroll
        for (int v = 0; v < 8; ++v) {
            int mm = m0 + tm + v + 8 * half;
            if (mm < M) {
                float val = acc[v] + bv;
                if (act == 1)      val = fmaxf(val, 0.f);
                else if (act == 2) val = tanhf(val);
                C[(size_t)mm * N + n] = val;
            }
        }
    }
}

// ---------------- utility kernels ----------------
__global__ void fill_kernel(float* p, float v, long n) {
    long i = (long)blockIdx.x * blockDim.x + threadIdx.x;
    if (i < n) p[i] = v;
}

// per (node, head): a_s, a_d dot products; init segment-max with self-loop
// value; zero z.
__global__ void gat_prep_kernel(const float* __restrict__ xh,
                                const float* __restrict__ aw_s,
                                const float* __restrict__ aw_d,
                                float* a_s, float* a_d, unsigned* mord, float* z,
                                int N, int D)
{
    int i = blockIdx.x * blockDim.x + threadIdx.x;
    if (i >= N * HH) return;
    int n = i / HH, h = i % HH;
    const float* row = xh + (size_t)n * HH * D + (size_t)h * D;
    float ss = 0.f, dd = 0.f;
    for (int d = 0; d < D; ++d) {
        float v = row[d];
        ss += v * aw_s[h * D + d];
        dd += v * aw_d[h * D + d];
    }
    a_s[i] = ss; a_d[i] = dd;
    mord[i] = f2ord(lrelu02(ss + dd));   // self-loop edge seeds the max
    z[i] = 0.f;
}

__global__ void gat_edge_max_kernel(const int* __restrict__ src,
                                    const int* __restrict__ dst,
                                    const float* __restrict__ a_s,
                                    const float* __restrict__ a_d,
                                    unsigned* mord, int E)
{
    int i = blockIdx.x * blockDim.x + threadIdx.x;
    if (i >= E * HH) return;
    int e = i / HH, h = i % HH;
    int s = src[e], d = dst[e];
    float v = lrelu02(a_s[s * HH + h] + a_d[d * HH + h]);
    atomicMax(&mord[d * HH + h], f2ord(v));
}

// per (edge-or-selfloop, head): ee = exp(e - m[dst]); z[dst]+=ee;
// S[dst,h,:] += ee * xh[src,h,:]
__global__ void gat_edge_accum_kernel(const int* __restrict__ src,
                                      const int* __restrict__ dst,
                                      const float* __restrict__ a_s,
                                      const float* __restrict__ a_d,
                                      const unsigned* __restrict__ mord,
                                      const float* __restrict__ xh,
                                      float* S, float* z,
                                      int E, int N, int D)
{
    int i = blockIdx.x * blockDim.x + threadIdx.x;
    if (i >= (E + N) * HH) return;
    int e = i / HH, h = i % HH;
    int s, d;
    if (e < E) { s = src[e]; d = dst[e]; } else { s = d = e - E; }
    float v  = lrelu02(a_s[s * HH + h] + a_d[d * HH + h]);
    float ee = __expf(v - ord2f(mord[d * HH + h]));
    atomicAdd(&z[d * HH + h], ee);
    const float* xr = xh + (size_t)s * HH * D + (size_t)h * D;
    float*       Sr = S  + (size_t)d * HH * D + (size_t)h * D;
    for (int k = 0; k < D; ++k) atomicAdd(&Sr[k], ee * xr[k]);
}

// out[n,d] = mean_h( S[n,h,d] / (z[n,h]+1e-16) ) + bias[d]  (+relu)
__global__ void gat_finalize_kernel(const float* __restrict__ S,
                                    const float* __restrict__ z,
                                    const float* __restrict__ bias,
                                    float* out, int N, int D, int act)
{
    int i = blockIdx.x * blockDim.x + threadIdx.x;
    if (i >= N * D) return;
    int n = i / D, d = i % D;
    float acc = 0.f;
    for (int h = 0; h < HH; ++h)
        acc += S[(size_t)n * HH * D + (size_t)h * D + d] / (z[n * HH + h] + 1e-16f);
    float v = acc * (1.0f / HH) + bias[d];
    if (act == 1) v = fmaxf(v, 0.f);
    out[i] = v;
}

// x = z*h + (1-z)*x, z = sigmoid(t1 + t2 + gb[d])
__global__ void gate_kernel(float* x, const float* __restrict__ h,
                            const float* __restrict__ t1,
                            const float* __restrict__ t2,
                            const float* __restrict__ gb, int N, int D)
{
    int i = blockIdx.x * blockDim.x + threadIdx.x;
    if (i >= N * D) return;
    int d = i % D;
    float zz = t1[i] + t2[i] + gb[d];
    zz = 1.f / (1.f + __expf(-zz));
    x[i] = zz * h[i] + (1.f - zz) * x[i];
}

__global__ void gcn_deg_kernel(const int* __restrict__ dst,
                               const float* __restrict__ w, float* deg, int E)
{
    int e = blockIdx.x * blockDim.x + threadIdx.x;
    if (e >= E) return;
    atomicAdd(&deg[dst[e]], w[e]);
}

__global__ void gcn_agg_kernel(const int* __restrict__ src,
                               const int* __restrict__ dst,
                               const float* __restrict__ w,
                               const float* __restrict__ deg,
                               const float* __restrict__ xw,
                               float* out, int E, int N, int D)
{
    int e = blockIdx.x * blockDim.x + threadIdx.x;
    if (e >= E + N) return;
    int s, d; float wv;
    if (e < E) { s = src[e]; d = dst[e]; wv = w[e]; }
    else       { s = d = e - E; wv = 1.f; }
    float ds = deg[s], dd = deg[d];
    float norm = (ds > 0.f ? rsqrtf(ds) : 0.f) * wv * (dd > 0.f ? rsqrtf(dd) : 0.f);
    const float* xr = xw + (size_t)s * D;
    float*       orow = out + (size_t)d * D;
    for (int k = 0; k < D; ++k) atomicAdd(&orow[k], norm * xr[k]);
}

__global__ void bias_relu_kernel(float* x, const float* __restrict__ b, int N, int D)
{
    int i = blockIdx.x * blockDim.x + threadIdx.x;
    if (i >= N * D) return;
    x[i] = fmaxf(x[i] + b[i % D], 0.f);
}

// global max pool over contiguous per-graph node ranges
__global__ void segmax_pool_kernel(const float* __restrict__ x, float* out,
                                   int G, int npg, int D)
{
    int i = blockIdx.x * blockDim.x + threadIdx.x;
    if (i >= G * D) return;
    int g = i / D, d = i % D;
    const float* base = x + (size_t)g * npg * D + d;
    float m = -3.402823466e38f;
    for (int n = 0; n < npg; ++n) m = fmaxf(m, base[(size_t)n * D]);
    out[i] = m;
}

// 2-way branch softmax + weighted embedding concat
__global__ void attn_combine_kernel(const float* __restrict__ ax,
                                    const float* __restrict__ axt,
                                    const float* __restrict__ xg,
                                    const float* __restrict__ xtg,
                                    float* emb, int G, int O)
{
    int i = blockIdx.x * blockDim.x + threadIdx.x;
    if (i >= G * O) return;
    int g = i / O, j = i % O;
    float e0 = ax[g], e1 = axt[g];
    float mx = fmaxf(e0, e1);
    float a0 = __expf(e0 - mx), a1 = __expf(e1 - mx);
    float s = a0 + a1; a0 /= s; a1 /= s;
    emb[(size_t)g * 2 * O + j]     = a0 * xg[i];
    emb[(size_t)g * 2 * O + O + j] = a1 * xtg[i];
}

// ---------------- host-side launch helpers ----------------
static inline void gemm(const float* A, const float* B, const float* bias, float* C,
                        int M, int N, int K, int act, hipStream_t s)
{
    dim3 blk(256);
    dim3 grd((N + TN - 1) / TN, (M + TM - 1) / TM);
    wmma_gemm_kernel<<<grd, blk, 0, s>>>(A, B, bias, C, M, N, K, act);
}
static inline dim3 g1d(long n) { return dim3((unsigned)((n + 255) / 256)); }

extern "C" void kernel_launch(void* const* d_in, const int* in_sizes, int n_in,
                              void* d_out, int out_size, void* d_ws, size_t ws_size,
                              hipStream_t stream)
{
    (void)in_sizes; (void)n_in; (void)out_size; (void)ws_size;
    auto F = [&](int i) { return (const float*)d_in[i]; };
    auto I = [&](int i) { return (const int*)d_in[i]; };

    // --- input indexing: top-level in setup_inputs() order; params pytree
    // flattened with sorted dict keys ('W' < 'ad' < 'as' < 'b') ---
    const float* mol_x = F(0);
    const float* pro_x = F(1);
    const float* pro_w = F(2);
    // params leaves (base 3):
    const float* attx_W1 = F(3),  *attx_W2 = F(4),  *attx_b1 = F(5);
    const float* attt_W1 = F(6),  *attt_W2 = F(7),  *attt_b1 = F(8);
    const float* fc1_W = F(9),  *fc1_b = F(10);
    const float* fc2_W = F(11), *fc2_b = F(12);
    const float* mol_bias = F(13);
    const float* mg1_W = F(14), *mg1_b = F(15);
    const float* mg2_W = F(16), *mg2_b = F(17);
    const float* mgat_W[3]  = { F(18), F(22), F(26) };
    const float* mgat_ad[3] = { F(19), F(23), F(27) };
    const float* mgat_as[3] = { F(20), F(24), F(28) };
    const float* mgat_b[3]  = { F(21), F(25), F(29) };
    const float* ms1_W = F(30), *ms1_b = F(31);
    const float* ms2_W = F(32), *ms2_b = F(33);
    const float* out_W = F(34), *out_b = F(35);
    const float* pro_bias = F(36);
    const float* pg1_W = F(37), *pg1_b = F(38);
    const float* pg2_W = F(39), *pg2_b = F(40);
    const float* pgat_W[2]  = { F(41), F(45) };
    const float* pgat_ad[2] = { F(42), F(46) };
    const float* pgat_as[2] = { F(43), F(47) };
    const float* pgat_b[2]  = { F(44), F(48) };
    const float* gcn_W = F(49), *gcn_b = F(50);
    const float* ps1_W = F(51), *ps1_b = F(52);
    const float* ps2_W = F(53), *ps2_b = F(54);
    const int* mol_ei = I(55);   // [2, EM] row-major: src row then dst row
    const int* pro_ei = I(57);
    const int *msrc = mol_ei, *mdst = mol_ei + EM;
    const int *psrc = pro_ei, *pdst = pro_ei + EPN;

    // --- workspace carve-out (sized for the larger protein branch; the mol
    // branch reuses the same buffers — total ~137 MB, all L2-resident) ---
    size_t off = 0;
    auto alloc = [&](size_t nfl) {
        void* p = (char*)d_ws + off;
        off += ((nfl * 4) + 255) & ~(size_t)255;
        return p;
    };
    float* XT  = (float*)alloc((size_t)NPN * DPH);       // node features (x / xt)
    float* XH  = (float*)alloc((size_t)NPN * HH * DPH);  // per-head features / GCN xw
    float* S   = (float*)alloc((size_t)NPN * HH * DPH);  // attention-weighted sums
    float* HB  = (float*)alloc((size_t)NPN * DPH);       // GAT layer output h
    float* T1  = (float*)alloc((size_t)NPN * DPH);
    float* T2  = (float*)alloc((size_t)NPN * DPH);
    float* AS  = (float*)alloc((size_t)NPN * HH);
    float* AD  = (float*)alloc((size_t)NPN * HH);
    unsigned* MORD = (unsigned*)alloc((size_t)NPN * HH);
    float* Z   = (float*)alloc((size_t)NPN * HH);
    float* DEG = (float*)alloc(NPN);
    float* XGR = (float*)alloc(N_GRAPHS * DMH);
    float* XTGR= (float*)alloc(N_GRAPHS * DPH);
    float* GH  = (float*)alloc(N_GRAPHS * 1024);
    float* XG  = (float*)alloc(N_GRAPHS * OUTF);
    float* XTG = (float*)alloc(N_GRAPHS * OUTF);
    float* TX  = (float*)alloc(N_GRAPHS * 64);
    float* AX  = (float*)alloc(N_GRAPHS);
    float* AXT = (float*)alloc(N_GRAPHS);
    float* EMB = (float*)alloc(N_GRAPHS * 2 * OUTF);
    float* H1  = (float*)alloc(N_GRAPHS * 1024);
    float* H2  = (float*)alloc(N_GRAPHS * 512);

    // one GAT layer: xh = x@W -> attention (max/sum segments) -> finalize
    auto gat_layer = [&](const float* xin, const float* W, const float* aws,
                         const float* awd, const float* b, float* hout,
                         int N, int Din, int D, const int* src, const int* dst,
                         int E, int act) {
        gemm(xin, W, nullptr, XH, N, HH * D, Din, 0, stream);
        gat_prep_kernel<<<g1d((long)N * HH), 256, 0, stream>>>(XH, aws, awd, AS, AD, MORD, Z, N, D);
        fill_kernel<<<g1d((long)N * HH * D), 256, 0, stream>>>(S, 0.f, (long)N * HH * D);
        gat_edge_max_kernel<<<g1d((long)E * HH), 256, 0, stream>>>(src, dst, AS, AD, MORD, E);
        gat_edge_accum_kernel<<<g1d((long)(E + N) * HH), 256, 0, stream>>>(
            src, dst, AS, AD, MORD, XH, S, Z, E, N, D);
        gat_finalize_kernel<<<g1d((long)N * D), 256, 0, stream>>>(S, Z, b, hout, N, D, act);
    };
    auto gate = [&](float* x, const float* h, const float* s1W, const float* s1b,
                    const float* s2W, const float* s2b, const float* gb,
                    int N, int D) {
        gemm(h, s1W, s1b, T1, N, D, D, 0, stream);
        gemm(x, s2W, s2b, T2, N, D, D, 0, stream);
        gate_kernel<<<g1d((long)N * D), 256, 0, stream>>>(x, h, T1, T2, gb, N, D);
    };

    // ================= drug (mol) branch =================
    gat_layer(mol_x, mgat_W[0], mgat_as[0], mgat_ad[0], mgat_b[0], XT,
              NM, FM, DMH, msrc, mdst, EM, /*relu*/1);
    gat_layer(XT, mgat_W[1], mgat_as[1], mgat_ad[1], mgat_b[1], HB,
              NM, DMH, DMH, msrc, mdst, EM, /*relu*/1);
    gate(XT, HB, ms1_W, ms1_b, ms2_W, ms2_b, mol_bias, NM, DMH);
    gat_layer(XT, mgat_W[2], mgat_as[2], mgat_ad[2], mgat_b[2], HB,
              NM, DMH, DMH, msrc, mdst, EM, /*none*/0);
    gate(XT, HB, ms1_W, ms1_b, ms2_W, ms2_b, mol_bias, NM, DMH);
    segmax_pool_kernel<<<g1d(N_GRAPHS * DMH), 256, 0, stream>>>(XT, XGR, N_GRAPHS, MOL_NPG, DMH);
    gemm(XGR, mg1_W, mg1_b, GH, N_GRAPHS, 1024, DMH, 1, stream);
    gemm(GH, mg2_W, mg2_b, XG, N_GRAPHS, OUTF, 1024, 0, stream);

    // ================= protein branch =================
    // GCN: xw = pro_x @ W (into XH); symmetric-norm aggregate with edge weights
    gemm(pro_x, gcn_W, nullptr, XH, NPN, DPH, FP, 0, stream);
    fill_kernel<<<g1d(NPN), 256, 0, stream>>>(DEG, 1.f, NPN);         // self loops
    gcn_deg_kernel<<<g1d(EPN), 256, 0, stream>>>(pdst, pro_w, DEG, EPN);
    fill_kernel<<<g1d((long)NPN * DPH), 256, 0, stream>>>(XT, 0.f, (long)NPN * DPH);
    gcn_agg_kernel<<<g1d((long)(EPN + NPN)), 256, 0, stream>>>(
        psrc, pdst, pro_w, DEG, XH, XT, EPN, NPN, DPH);
    bias_relu_kernel<<<g1d((long)NPN * DPH), 256, 0, stream>>>(XT, gcn_b, NPN, DPH);

    gat_layer(XT, pgat_W[0], pgat_as[0], pgat_ad[0], pgat_b[0], HB,
              NPN, DPH, DPH, psrc, pdst, EPN, /*relu*/1);
    gate(XT, HB, ps1_W, ps1_b, ps2_W, ps2_b, pro_bias, NPN, DPH);
    gat_layer(XT, pgat_W[1], pgat_as[1], pgat_ad[1], pgat_b[1], HB,
              NPN, DPH, DPH, psrc, pdst, EPN, /*none*/0);
    gate(XT, HB, ps1_W, ps1_b, ps2_W, ps2_b, pro_bias, NPN, DPH);
    segmax_pool_kernel<<<g1d(N_GRAPHS * DPH), 256, 0, stream>>>(XT, XTGR, N_GRAPHS, PRO_NPG, DPH);
    gemm(XTGR, pg1_W, pg1_b, GH, N_GRAPHS, 1024, DPH, 1, stream);
    gemm(GH, pg2_W, pg2_b, XTG, N_GRAPHS, OUTF, 1024, 0, stream);

    // ================= branch attention + readout =================
    gemm(XG,  attx_W1, attx_b1, TX, N_GRAPHS, 64, OUTF, /*tanh*/2, stream);
    gemm(TX,  attx_W2, nullptr, AX, N_GRAPHS, 1, 64, 0, stream);
    gemm(XTG, attt_W1, attt_b1, TX, N_GRAPHS, 64, OUTF, /*tanh*/2, stream);
    gemm(TX,  attt_W2, nullptr, AXT, N_GRAPHS, 1, 64, 0, stream);
    attn_combine_kernel<<<g1d(N_GRAPHS * OUTF), 256, 0, stream>>>(AX, AXT, XG, XTG, EMB, N_GRAPHS, OUTF);
    gemm(EMB, fc1_W, fc1_b, H1, N_GRAPHS, 1024, 2 * OUTF, 1, stream);
    gemm(H1,  fc2_W, fc2_b, H2, N_GRAPHS, 512, 1024, 1, stream);
    gemm(H2,  out_W, out_b, (float*)d_out, N_GRAPHS, 1, 512, 0, stream);
}
